// AKDN_50775103373668
// MI455X (gfx1250) — matrix operand order repaired and checked by reading, not compiled, not run on hardware
//
#include <hip/hip_runtime.h>
#include <hip/hip_bf16.h>

// ---------------------------------------------------------------------------
// KGAT-style model, MI455X (gfx1250, wave32, WMMA) implementation.
// Dominant cost: 2M-edge scatter/gather (~4 GB HBM traffic) -> memory bound.
// GEMMs mapped to v_wmma_f32_16x16x32_f16 (f16 in, f32 acc).
// Weight staging into LDS uses CDNA5 async global->LDS DMA (ASYNCcnt).
// ---------------------------------------------------------------------------

#define DD    64
#define KNB   16
#define NUc   100000
#define NIc   50000
#define NEc   200000
#define NRc   64
#define EDGESc 2000000
#define Bc    4096
#define Lc    2
#define REGc  1e-4f

typedef __attribute__((ext_vector_type(16))) _Float16 v16h;
typedef __attribute__((ext_vector_type(8)))  _Float16 v8h;
typedef __attribute__((ext_vector_type(8)))  float    v8f;

// --- CDNA5 async global->LDS DMA (tracked by ASYNCcnt) ---------------------
// LDS destination byte offset = low 32 bits of generic shared-space address
// (ISA 10.2, LDS aperture).
__device__ __forceinline__ void async_load_b128(void* lds_ptr, const float* gaddr) {
    unsigned lds_off = (unsigned)(size_t)lds_ptr;
    asm volatile("global_load_async_to_lds_b128 %0, %1, off"
                 :: "v"(lds_off), "v"(gaddr) : "memory");
}
__device__ __forceinline__ void wait_async0() {
    asm volatile("s_wait_asynccnt 0" ::: "memory");
}

// Build a 16-half A-fragment chunk from an fp32 row in global memory.
// Layout (16-bit A 16x32): lane half h: elems 0..7 -> d = 32c + 8h + e,
// elems 8..15 -> d = 32c + 16 + 8h + (e-8).
__device__ __forceinline__ v16h pack_rowchunk(const float* __restrict__ row, int c, int h) {
    v16h r;
#pragma unroll
    for (int e = 0; e < 8; ++e) r[e]     = (_Float16)row[32 * c + 8 * h + e];
#pragma unroll
    for (int e = 0; e < 8; ++e) r[8 + e] = (_Float16)row[32 * c + 16 + 8 * h + e];
    return r;
}

// ---------------------------------------------------------------------------
// init / utility kernels
// ---------------------------------------------------------------------------
__global__ void __launch_bounds__(256) akdn_copy2(const float* __restrict__ src,
                                                  float* __restrict__ d1,
                                                  float* __restrict__ d2, int n) {
    int i = blockIdx.x * 256 + threadIdx.x;
    if (i < n) { float v = src[i]; d1[i] = v; d2[i] = v; }
}

__global__ void __launch_bounds__(256) akdn_zero(float* __restrict__ p, int n) {
    int i = blockIdx.x * 256 + threadIdx.x;
    if (i < n) p[i] = 0.0f;
}

__global__ void __launch_bounds__(256) akdn_u_update(const float* __restrict__ agg_u,
                                                     float* __restrict__ u_cur,
                                                     float* __restrict__ u_sum, int n) {
    int i = blockIdx.x * 256 + threadIdx.x;
    if (i < n) { float v = agg_u[i]; u_cur[i] = v; u_sum[i] += v; }
}

// ---------------------------------------------------------------------------
// KG attention: per item i: hv = v * i_cur[i]; proj = hv @ w_eff^T + b;
// score = leaky_relu(sum(r*proj)); alpha = softmax_k; kg_item = alpha @ v.
// One wave per item; proj GEMM is 16x64x64 -> 8 v_wmma_f32_16x16x32_f16.
// Wk weights (32 KB) staged via async global->LDS DMA, overlapped with hv
// construction, then converted to f16 w_eff in LDS.
// ---------------------------------------------------------------------------
__global__ void __launch_bounds__(256) akdn_attention(
    const float* __restrict__ i_cur, const int* __restrict__ kg_ent,
    const int* __restrict__ kg_rel, const float* __restrict__ ent_emb,
    const float* __restrict__ rel_emb, const float* __restrict__ Wk_w_l,
    const float* __restrict__ Wk_b_l, float* __restrict__ kg_item) {
    __shared__ alignas(16) float stage[DD * 2 * DD];  // raw Wk[l] (64x128 f32)
    __shared__ alignas(32) _Float16 weff[DD * DD];    // w_eff[o][d], f16
    __shared__ float wkb[DD];
    __shared__ alignas(32) _Float16 hv[8][KNB * DD];  // per-wave 16x64 f16 tile

    int tid = threadIdx.x;
    // Kick off async DMA of the layer's Wk weights into LDS (8 x B128 each).
#pragma unroll
    for (int b = 0; b < 8; ++b) {
        int q = tid + b * 256;  // float4 index, 2048 total
        async_load_b128(&stage[q * 4], Wk_w_l + q * 4);
    }

    // Overlap: build the per-wave hv tile while the DMA is in flight.
    int wave = tid >> 5, lane = tid & 31;
    int item = blockIdx.x * 8 + wave;  // NI divisible by 8
    _Float16* hvw = hv[wave];
    int d0 = lane * 2;
    float ic0 = i_cur[(size_t)item * DD + d0];
    float ic1 = i_cur[(size_t)item * DD + d0 + 1];
    const int* ent_row = kg_ent + (size_t)item * KNB;
    const int* rel_row = kg_rel + (size_t)item * KNB;
#pragma unroll
    for (int k = 0; k < KNB; ++k) {
        const float* vrow = ent_emb + (size_t)ent_row[k] * DD;
        float2 vv = *(const float2*)(vrow + d0);
        hvw[k * DD + d0]     = (_Float16)(vv.x * ic0);
        hvw[k * DD + d0 + 1] = (_Float16)(vv.y * ic1);
    }
    wait_async0();
    __syncthreads();

    // Convert staged f32 weights to f16 w_eff = Wk[:, :D] + Wk[:, D:].
    for (int idx = tid; idx < DD * DD; idx += 256) {
        int o = idx >> 6, d = idx & 63;
        weff[idx] = (_Float16)(stage[o * 2 * DD + d] + stage[o * 2 * DD + DD + d]);
    }
    if (tid < DD) wkb[tid] = Wk_b_l[tid];
    __syncthreads();

    int m = lane & 15, h = lane >> 4;
    // A fragments: rows = K-neighbors, k-dim = embedding d (two chunks of 32)
    v16h a[2];
#pragma unroll
    for (int c = 0; c < 2; ++c) {
        v8h lo = *(const v8h*)(hvw + m * DD + 32 * c + 8 * h);
        v8h hi = *(const v8h*)(hvw + m * DD + 32 * c + 16 + 8 * h);
        a[c] = __builtin_shufflevector(lo, hi, 0, 1, 2, 3, 4, 5, 6, 7, 8, 9, 10,
                                       11, 12, 13, 14, 15);
    }

    float part[8];
#pragma unroll
    for (int j = 0; j < 8; ++j) part[j] = 0.0f;

#pragma unroll
    for (int t = 0; t < 4; ++t) {
        int col = t * 16 + m;  // output dim handled by this lane in this tile
        float bias = wkb[col];
        v8f acc = {bias, bias, bias, bias, bias, bias, bias, bias};
#pragma unroll
        for (int c = 0; c < 2; ++c) {
            v8h blo = *(const v8h*)(weff + col * DD + 32 * c + 8 * h);
            v8h bhi = *(const v8h*)(weff + col * DD + 32 * c + 16 + 8 * h);
            v16h b = __builtin_shufflevector(blo, bhi, 0, 1, 2, 3, 4, 5, 6, 7, 8, 9, 10, 11, 12, 13, 14, 15);
            acc = __builtin_amdgcn_wmma_f32_16x16x32_f16(false, a[c], false, b,
                                                         (short)0, acc, false, false);
        }
        // score partials: D layout VGPR j -> row j (lanes 0-15) / j+8 (16-31)
#pragma unroll
        for (int j = 0; j < 8; ++j) {
            int row = h ? (j + 8) : j;
            float rv = rel_emb[(size_t)rel_row[row] * DD + col];
            part[j] += rv * acc[j];
        }
    }
    // reduce over the 16 columns in this half-wave
#pragma unroll
    for (int off = 1; off < 16; off <<= 1) {
#pragma unroll
        for (int j = 0; j < 8; ++j) part[j] += __shfl_xor(part[j], off, 32);
    }
    float s[16];
#pragma unroll
    for (int j = 0; j < 8; ++j) {
        float other = __shfl_xor(part[j], 16, 32);
        s[j]     = h ? other : part[j];
        s[j + 8] = h ? part[j] : other;
    }
    // leaky_relu(0.2) + softmax over K=16 (redundant per lane)
    float mx = -3.4e38f;
#pragma unroll
    for (int k = 0; k < 16; ++k) {
        s[k] = (s[k] > 0.0f) ? s[k] : 0.2f * s[k];
        mx = fmaxf(mx, s[k]);
    }
    float den = 0.0f;
#pragma unroll
    for (int k = 0; k < 16; ++k) { s[k] = expf(s[k] - mx); den += s[k]; }
    float inv = 1.0f / den;
    // kg_item[d0..d0+1] = sum_k alpha[k] * v[k][d]
    float o0 = 0.0f, o1 = 0.0f;
#pragma unroll
    for (int k = 0; k < 16; ++k) {
        const float* vrow = ent_emb + (size_t)ent_row[k] * DD;
        float2 vv = *(const float2*)(vrow + d0);
        float al = s[k] * inv;
        o0 += al * vv.x;
        o1 += al * vv.y;
    }
    kg_item[(size_t)item * DD + d0]     = o0;
    kg_item[(size_t)item * DD + d0 + 1] = o1;
}

// ---------------------------------------------------------------------------
// Edge aggregation: agg_i[ei] += norm*u_cur[eu]; agg_u[eu] += norm*i_cur[ei]
// 16 threads/edge, float4 gathers, f32 global atomics. Memory-bound roofline.
// ---------------------------------------------------------------------------
__global__ void __launch_bounds__(256) akdn_edges(
    const float* __restrict__ u_cur, const float* __restrict__ i_cur,
    const int* __restrict__ edge_u, const int* __restrict__ edge_i,
    const float* __restrict__ norm, float* __restrict__ agg_u,
    float* __restrict__ agg_i) {
    int tid = blockIdx.x * 256 + threadIdx.x;  // EDGES*16 threads exactly
    int e = tid >> 4;
    int d0 = (tid & 15) * 4;
    int eu = edge_u[e], ei = edge_i[e];
    float w = norm[e];
    float4 uc = *(const float4*)(u_cur + (size_t)eu * DD + d0);
    float4 ic = *(const float4*)(i_cur + (size_t)ei * DD + d0);
    float* ai = agg_i + (size_t)ei * DD + d0;
    float* au = agg_u + (size_t)eu * DD + d0;
    atomicAdd(ai + 0, w * uc.x); atomicAdd(ai + 1, w * uc.y);
    atomicAdd(ai + 2, w * uc.z); atomicAdd(ai + 3, w * uc.w);
    atomicAdd(au + 0, w * ic.x); atomicAdd(au + 1, w * ic.y);
    atomicAdd(au + 2, w * ic.z); atomicAdd(au + 3, w * ic.w);
}

// ---------------------------------------------------------------------------
// Gate: gate = sigmoid(kg@Wa^T + agg@Wb^T + ba + bb);
// i_cur = gate*kg + (1-gate)*agg; i_sum += i_cur.
// One wave per 16 items; 16 v_wmma per wave (2 GEMMs into one accumulator).
// Wa/Wb (32 KB) staged via async global->LDS DMA.
// ---------------------------------------------------------------------------
__global__ void __launch_bounds__(256) akdn_gate(
    const float* __restrict__ kg_item, const float* __restrict__ agg_i,
    const float* __restrict__ Wa_w_l, const float* __restrict__ Wa_b_l,
    const float* __restrict__ Wb_w_l, const float* __restrict__ Wb_b_l,
    float* __restrict__ i_cur, float* __restrict__ i_sum) {
    __shared__ alignas(16) float stage[2 * DD * DD];  // Wa | Wb raw f32
    __shared__ alignas(32) _Float16 wa[DD * DD];
    __shared__ alignas(32) _Float16 wb[DD * DD];
    __shared__ float bias[DD];

    int tid = threadIdx.x;
    // Async DMA both weight matrices into LDS (8 x B128 per thread).
#pragma unroll
    for (int b = 0; b < 8; ++b) {
        int q = tid + b * 256;  // float4 index, 2048 total (1024 Wa + 1024 Wb)
        const float* g = (q < 1024) ? (Wa_w_l + q * 4) : (Wb_w_l + (q - 1024) * 4);
        async_load_b128(&stage[q * 4], g);
    }

    int wave = tid >> 5, lane = tid & 31;
    int m = lane & 15, h = lane >> 4;
    int base = (blockIdx.x * 8 + wave) * 16;
    int it_a = base + m;
    int it_c = (it_a < NIc) ? it_a : (NIc - 1);  // clamp loads; stores guarded

    // Overlap: build A fragments from kg_item/agg_i while the DMA runs.
    const float* kr = kg_item + (size_t)it_c * DD;
    const float* ar = agg_i + (size_t)it_c * DD;
    v16h akg[2], aag[2];
#pragma unroll
    for (int c = 0; c < 2; ++c) {
        akg[c] = pack_rowchunk(kr, c, h);
        aag[c] = pack_rowchunk(ar, c, h);
    }
    wait_async0();
    __syncthreads();

    for (int idx = tid; idx < DD * DD; idx += 256) {
        wa[idx] = (_Float16)stage[idx];
        wb[idx] = (_Float16)stage[DD * DD + idx];
    }
    if (tid < DD) bias[tid] = Wa_b_l[tid] + Wb_b_l[tid];
    __syncthreads();

#pragma unroll
    for (int t = 0; t < 4; ++t) {
        int col = t * 16 + m;
        float bsv = bias[col];
        v8f acc = {bsv, bsv, bsv, bsv, bsv, bsv, bsv, bsv};
#pragma unroll
        for (int c = 0; c < 2; ++c) {
            v8h walo = *(const v8h*)(wa + col * DD + 32 * c + 8 * h);
            v8h wahi = *(const v8h*)(wa + col * DD + 32 * c + 16 + 8 * h);
            v16h bwa = __builtin_shufflevector(walo, wahi, 0, 1, 2, 3, 4, 5, 6, 7, 8, 9, 10, 11, 12, 13, 14, 15);
            acc = __builtin_amdgcn_wmma_f32_16x16x32_f16(false, akg[c], false, bwa,
                                                         (short)0, acc, false, false);
            v8h wblo = *(const v8h*)(wb + col * DD + 32 * c + 8 * h);
            v8h wbhi = *(const v8h*)(wb + col * DD + 32 * c + 16 + 8 * h);
            v16h bwb = __builtin_shufflevector(wblo, wbhi, 0, 1, 2, 3, 4, 5, 6, 7, 8, 9, 10, 11, 12, 13, 14, 15);
            acc = __builtin_amdgcn_wmma_f32_16x16x32_f16(false, aag[c], false, bwb,
                                                         (short)0, acc, false, false);
        }
#pragma unroll
        for (int j = 0; j < 8; ++j) {
            int it = base + (h ? (j + 8) : j);
            if (it < NIc) {
                float g = 1.0f / (1.0f + expf(-acc[j]));
                float kv = kg_item[(size_t)it * DD + col];
                float av = agg_i[(size_t)it * DD + col];
                float nv = g * kv + (1.0f - g) * av;
                i_cur[(size_t)it * DD + col] = nv;
                i_sum[(size_t)it * DD + col] += nv;
            }
        }
    }
}

// ---------------------------------------------------------------------------
// BPR loss + L2: one wave per batch row, shfl reduction, scalar atomics.
// ---------------------------------------------------------------------------
__global__ void __launch_bounds__(256) akdn_loss(
    const float* __restrict__ u_sum, const float* __restrict__ i_sum,
    const int* __restrict__ user, const int* __restrict__ pos,
    const int* __restrict__ neg, float* __restrict__ scal) {
    int gtid = blockIdx.x * 256 + threadIdx.x;
    int b = gtid >> 5;  // exactly B waves
    int lane = gtid & 31;
    const float* ue = u_sum + (size_t)user[b] * DD;
    const float* pe = i_sum + (size_t)pos[b] * DD;
    const float* ne = i_sum + (size_t)neg[b] * DD;
    int d0 = lane * 2;
    float2 u = *(const float2*)(ue + d0);
    float2 p = *(const float2*)(pe + d0);
    float2 n = *(const float2*)(ne + d0);
    float ps = u.x * p.x + u.y * p.y;
    float ns = u.x * n.x + u.y * n.y;
    float l2 = u.x * u.x + u.y * u.y + p.x * p.x + p.y * p.y + n.x * n.x + n.y * n.y;
#pragma unroll
    for (int off = 1; off < 32; off <<= 1) {
        ps += __shfl_xor(ps, off, 32);
        ns += __shfl_xor(ns, off, 32);
        l2 += __shfl_xor(l2, off, 32);
    }
    if (lane == 0) {
        float x = ps - ns;
        float sig = 1.0f / (1.0f + expf(-x));
        atomicAdd(&scal[0], -logf(sig + 1e-10f));
        atomicAdd(&scal[1], l2);
    }
}

__global__ void akdn_finalize(const float* __restrict__ scal, float* __restrict__ out) {
    out[0] = scal[0] / (float)Bc + REGc * (scal[1] / (float)Bc);
}

// ---------------------------------------------------------------------------
extern "C" void kernel_launch(void* const* d_in, const int* in_sizes, int n_in,
                              void* d_out, int out_size, void* d_ws, size_t ws_size,
                              hipStream_t stream) {
    const float* user_emb = (const float*)d_in[0];
    const float* item_emb = (const float*)d_in[1];
    const float* ent_emb  = (const float*)d_in[2];
    const float* rel_emb  = (const float*)d_in[3];
    const float* Wk_w = (const float*)d_in[4];
    const float* Wk_b = (const float*)d_in[5];
    const float* Wa_w = (const float*)d_in[6];
    const float* Wa_b = (const float*)d_in[7];
    const float* Wb_w = (const float*)d_in[8];
    const float* Wb_b = (const float*)d_in[9];
    const float* norm = (const float*)d_in[10];
    const int* user = (const int*)d_in[11];
    const int* pos  = (const int*)d_in[12];
    const int* neg  = (const int*)d_in[13];
    const int* eu   = (const int*)d_in[14];
    const int* ei   = (const int*)d_in[15];
    const int* kg_rel = (const int*)d_in[16];
    const int* kg_ent = (const int*)d_in[17];

    // workspace layout (floats); ~128 MB total
    float* ws = (float*)d_ws;
    float* u_cur = ws;
    float* u_sum = u_cur + (size_t)NUc * DD;
    float* i_cur = u_sum + (size_t)NUc * DD;
    float* i_sum = i_cur + (size_t)NIc * DD;
    float* agg_u = i_sum + (size_t)NIc * DD;
    float* agg_i = agg_u + (size_t)NUc * DD;
    float* scal  = agg_i + (size_t)NIc * DD;   // 2 scalars (padded to 64)
    float* kg_it = scal + 64;

    akdn_copy2<<<(NUc * DD) / 256, 256, 0, stream>>>(user_emb, u_cur, u_sum, NUc * DD);
    akdn_copy2<<<(NIc * DD) / 256, 256, 0, stream>>>(item_emb, i_cur, i_sum, NIc * DD);

    for (int l = 0; l < Lc; ++l) {
        int zn = NUc * DD + NIc * DD + 64;  // agg_u + agg_i + scal pad (contiguous)
        akdn_zero<<<(zn + 255) / 256, 256, 0, stream>>>(agg_u, zn);
        akdn_attention<<<NIc / 8, 256, 0, stream>>>(
            i_cur, kg_ent, kg_rel, ent_emb, rel_emb,
            Wk_w + (size_t)l * DD * 2 * DD, Wk_b + (size_t)l * DD, kg_it);
        akdn_edges<<<(EDGESc * 16) / 256, 256, 0, stream>>>(u_cur, i_cur, eu, ei,
                                                            norm, agg_u, agg_i);
        akdn_gate<<<(NIc + 127) / 128, 256, 0, stream>>>(
            kg_it, agg_i, Wa_w + (size_t)l * DD * DD, Wa_b + (size_t)l * DD,
            Wb_w + (size_t)l * DD * DD, Wb_b + (size_t)l * DD, i_cur, i_sum);
        akdn_u_update<<<(NUc * DD) / 256, 256, 0, stream>>>(agg_u, u_cur, u_sum,
                                                            NUc * DD);
    }
    akdn_loss<<<(Bc * 32) / 256, 256, 0, stream>>>(u_sum, i_sum, user, pos, neg, scal);
    akdn_finalize<<<1, 1, 0, stream>>>(scal, (float*)d_out);
}
